// LSTM_model_21509196218918
// MI455X (gfx1250) — matrix operand order repaired and checked by reading, not compiled, not run on hardware
//
#include <hip/hip_runtime.h>

typedef __attribute__((ext_vector_type(16))) __bf16 bf16x16;
typedef __attribute__((ext_vector_type(8)))  __bf16 bf16x8;
typedef __attribute__((ext_vector_type(8)))  float  f32x8;

namespace {

constexpr int T_STEPS  = 512;
constexpr int BATCH    = 1024;
constexpr int S_IN     = 64;
constexpr int HID      = 128;
constexpr int NGATE    = 512;   // 4*HID
constexpr int ACT      = 10;
constexpr int MTILE    = 16;    // batch rows per workgroup
constexpr int NTHREADS = 512;   // 16 waves (wave32) -> 4 waves/SIMD, 256-VGPR budget
constexpr int CVT_THREADS = 256;

// padded LDS row strides (elements) - keep 16B alignment + spread banks
constexpr int XPAD = 72;        // x tile / Wih1 rows (bf16): 144B = 9*16
constexpr int HPAD = 136;       // h tile / Whh1 rows (bf16): 272B = 17*16
constexpr int GPAD = 516;       // gates rows   (f32) : 2064B = 129*16
constexpr int FPAD = 132;       // fp32 rows    (f32) : 528B = 33*16

union AFrag { bf16x16 v; struct { bf16x8 lo, hi; } h; };

// A-matrix 16x32 bf16 layout: lanes 0-15 -> rows 0-15;
// lane half lhi selects contiguous K half (K=0..15 / 16..31).
__device__ __forceinline__ bf16x16 load_a(const __bf16* rowp, int kbase, int lhi) {
  AFrag a;
  a.h.lo = *(const bf16x8*)(rowp + kbase +     lhi * 16);
  a.h.hi = *(const bf16x8*)(rowp + kbase + 8 + lhi * 16);
  return a.v;
}

// B-matrix 32x16 bf16 layout: lane = column (lane&15); lanes 0-15 hold
// K=kbase..kbase+15, lanes 16-31 hold K=kbase+16..kbase+31, contiguous in W row.
__device__ __forceinline__ bf16x16 load_b(const __bf16* __restrict__ W, int Ktot,
                                          int col, int kbase, int lhi) {
  return *(const bf16x16*)(W + (size_t)col * Ktot + kbase + lhi * 16);
}

// Same B-fragment gather but from an LDS image with padded row stride.
__device__ __forceinline__ bf16x16 load_b_lds(const __bf16* Wlds, int stride,
                                              int col, int kbase, int lhi) {
  const __bf16* p = Wlds + col * stride + kbase + lhi * 16;
  AFrag b;
  b.h.lo = *(const bf16x8*)(p);
  b.h.hi = *(const bf16x8*)(p + 8);
  return b.v;
}

__device__ __forceinline__ float sigmoidf(float x) {
  return 1.f / (1.f + __expf(-x));
}

// branch-free, overflow-safe tanh: avoids ocml's divergent slow path
__device__ __forceinline__ float fast_tanhf(float x) {
  float a = __builtin_fabsf(x);
  float e = __expf(2.f * a);           // +inf for large a -> r -> 1
  float r = 1.f - 2.f / (e + 1.f);
  return __builtin_copysignf(r, x);
}

} // namespace

extern "C" __global__ void __launch_bounds__(CVT_THREADS)
cvt_f32_bf16(const float* __restrict__ src, __bf16* __restrict__ dst, int n) {
  int i = blockIdx.x * blockDim.x + threadIdx.x;
  if (i < n) dst[i] = (__bf16)src[i];
}

extern "C" __global__ void __launch_bounds__(NTHREADS)
lstm2_wmma(const float* __restrict__ x,
           const __bf16* __restrict__ Wih1, const __bf16* __restrict__ Whh1,
           const __bf16* __restrict__ Wih2, const __bf16* __restrict__ Whh2,
           const float* __restrict__ b_ih1, const float* __restrict__ b_hh1,
           const float* __restrict__ b_ih2, const float* __restrict__ b_hh2,
           const float* __restrict__ Wout,  const float* __restrict__ bout_g,
           float* __restrict__ out) {
  __shared__ __bf16 xbuf [MTILE][XPAD];
  __shared__ __bf16 h1buf[MTILE][HPAD];
  __shared__ __bf16 h2buf[MTILE][HPAD];
  __shared__ float  gates[MTILE][GPAD];
  __shared__ float  h2f  [MTILE][FPAD];
  __shared__ float  cbuf1[MTILE][FPAD];    // LSTM cell state (thread-private slots)
  __shared__ float  cbuf2[MTILE][FPAD];
  __shared__ float  wout_s[ACT][FPAD];
  __shared__ float  bias1[NGATE];
  __shared__ float  bias2[NGATE];
  __shared__ float  bout_s[16];
  __shared__ __bf16 wih1_s[NGATE][XPAD];   // 73.7KB LDS image of Wih1
  __shared__ __bf16 whh1_s[NGATE][HPAD];   // 139.3KB LDS image of Whh1

  const int tid  = threadIdx.x;
  const int lane = tid & 31;
  const int wave = tid >> 5;        // 0..15, two gate N-tiles each
  const int lrow = lane & 15;       // A row / B col / D col within tile
  const int lhi  = lane >> 4;       // lane half selector
  const int b0   = blockIdx.x * MTILE;
  const int nt0  = wave * 2;        // first of this wave's two N-tiles

  // ---- one-time setup ----
  if (tid < NGATE) {
    bias1[tid] = b_ih1[tid] + b_hh1[tid];
    bias2[tid] = b_ih2[tid] + b_hh2[tid];
  }
  for (int i = tid; i < ACT * HID; i += NTHREADS)
    wout_s[i / HID][i % HID] = Wout[i];
  if (tid < ACT) bout_s[tid] = bout_g[tid];
  for (int i = tid; i < MTILE * HID; i += NTHREADS) {
    h1buf[i / HID][i % HID] = (__bf16)0.f;
    h2buf[i / HID][i % HID] = (__bf16)0.f;
    cbuf1[i / HID][i % HID] = 0.f;
    cbuf2[i / HID][i % HID] = 0.f;
  }
  // stage Wih1 / Whh1 into LDS once (row-major, padded rows)
  for (int i = tid; i < NGATE * S_IN; i += NTHREADS)
    wih1_s[i >> 6][i & 63] = Wih1[i];
  for (int i = tid; i < NGATE * HID; i += NTHREADS)
    whh1_s[i >> 7][i & 127] = Whh1[i];

  // ---- register-resident weight B-fragments for layer 2 ----
  // 2 N-tiles x 8 fragments x 8 VGPRs = 128 VGPRs/wave, loaded from L2 once.
  bf16x16 f_ih2[2][4], f_hh2[2][4];
#pragma unroll
  for (int q = 0; q < 2; ++q) {
    const int bc = (nt0 + q) * 16 + lrow;
#pragma unroll
    for (int kt = 0; kt < HID / 32; ++kt) {
      f_ih2[q][kt] = load_b(Wih2, HID, bc, kt * 32, lhi);
      f_hh2[q][kt] = load_b(Whh2, HID, bc, kt * 32, lhi);
    }
  }

  const int er  = tid >> 5;         // elementwise row 0..15 (32 threads/row)
  const int ecb = (tid & 31) * 4;   // elementwise col base 0..124

  __syncthreads();

#pragma unroll 1
  for (int t = 0; t < T_STEPS; ++t) {
    // ---- stage x_t tile (fp32 -> bf16 LDS) ----
    {
      int r = tid >> 5;             // 0..15
      int s = (tid & 31) * 2;       // 0..62
      const float* xp = x + ((size_t)t * BATCH + b0 + r) * S_IN + s;
      float2 v = *(const float2*)xp;
      xbuf[r][s + 0] = (__bf16)v.x;
      xbuf[r][s + 1] = (__bf16)v.y;
      if (t + 1 < T_STEPS)
        __builtin_prefetch(xp + (size_t)BATCH * S_IN, 0, 1); // global_prefetch
    }
    __syncthreads();

    // ---- layer-1 gates = x_t @ Wih1^T + h1 @ Whh1^T (B operands from LDS) ----
#pragma unroll
    for (int q = 0; q < 2; ++q) {
      const int n0 = (nt0 + q) * 16;
      f32x8 acc = {};
      const __bf16* arow_x = &xbuf[lrow][0];
#pragma unroll
      for (int kt = 0; kt < S_IN / 32; ++kt)
        acc = __builtin_amdgcn_wmma_f32_16x16x32_bf16(
            false, load_a(arow_x, kt * 32, lhi),
            false, load_b_lds(&wih1_s[0][0], XPAD, n0 + lrow, kt * 32, lhi),
            (short)0, acc, false, false);
      const __bf16* arow_h = &h1buf[lrow][0];
#pragma unroll
      for (int kt = 0; kt < HID / 32; ++kt)
        acc = __builtin_amdgcn_wmma_f32_16x16x32_bf16(
            false, load_a(arow_h, kt * 32, lhi),
            false, load_b_lds(&whh1_s[0][0], HPAD, n0 + lrow, kt * 32, lhi),
            (short)0, acc, false, false);
#pragma unroll
      for (int v = 0; v < 8; ++v)
        gates[v + 8 * lhi][n0 + lrow] = acc[v];
    }
    __syncthreads();

    // ---- layer-1 elementwise (cell state in thread-private LDS slots) ----
#pragma unroll
    for (int j = 0; j < 4; ++j) {
      int c = ecb + j;
      float gi = gates[er][c]           + bias1[c];
      float gf = gates[er][HID + c]     + bias1[HID + c];
      float gg = gates[er][2 * HID + c] + bias1[2 * HID + c];
      float go = gates[er][3 * HID + c] + bias1[3 * HID + c];
      float ii = sigmoidf(gi);
      float ff = sigmoidf(gf);
      float g  = fast_tanhf(gg);
      float oo = sigmoidf(go);
      float cn = ff * cbuf1[er][c] + ii * g;
      cbuf1[er][c] = cn;
      h1buf[er][c] = (__bf16)(oo * fast_tanhf(cn));
    }
    __syncthreads();

    // ---- layer-2 gates = h1 @ Wih2^T + h2 @ Whh2^T (B operands in VGPRs) ----
#pragma unroll
    for (int q = 0; q < 2; ++q) {
      const int n0 = (nt0 + q) * 16;
      f32x8 acc = {};
      const __bf16* arow_h1 = &h1buf[lrow][0];
#pragma unroll
      for (int kt = 0; kt < HID / 32; ++kt)
        acc = __builtin_amdgcn_wmma_f32_16x16x32_bf16(
            false, load_a(arow_h1, kt * 32, lhi),
            false, f_ih2[q][kt], (short)0, acc, false, false);
      const __bf16* arow_h2 = &h2buf[lrow][0];
#pragma unroll
      for (int kt = 0; kt < HID / 32; ++kt)
        acc = __builtin_amdgcn_wmma_f32_16x16x32_bf16(
            false, load_a(arow_h2, kt * 32, lhi),
            false, f_hh2[q][kt], (short)0, acc, false, false);
#pragma unroll
      for (int v = 0; v < 8; ++v)
        gates[v + 8 * lhi][n0 + lrow] = acc[v];
    }
    __syncthreads();

    // ---- layer-2 elementwise (cell state in thread-private LDS slots) ----
#pragma unroll
    for (int j = 0; j < 4; ++j) {
      int c = ecb + j;
      float gi = gates[er][c]           + bias2[c];
      float gf = gates[er][HID + c]     + bias2[HID + c];
      float gg = gates[er][2 * HID + c] + bias2[2 * HID + c];
      float go = gates[er][3 * HID + c] + bias2[3 * HID + c];
      float ii = sigmoidf(gi);
      float ff = sigmoidf(gf);
      float g  = fast_tanhf(gg);
      float oo = sigmoidf(go);
      float cn = ff * cbuf2[er][c] + ii * g;
      cbuf2[er][c] = cn;
      float h = oo * fast_tanhf(cn);
      h2buf[er][c] = (__bf16)h;
      h2f[er][c]   = h;
    }
    __syncthreads();

    // ---- fused output projection mu = h2 @ Wout^T + b_out (A=10) ----
    if (tid < MTILE * ACT) {
      int r = tid & 15;
      int a = tid >> 4;
      float accp = bout_s[a];
      const float4* hp = (const float4*)&h2f[r][0];
      const float4* wp = (const float4*)&wout_s[a][0];
#pragma unroll 4
      for (int k = 0; k < HID / 4; ++k) {
        float4 hv = hp[k];
        float4 wv = wp[k];
        accp += hv.x * wv.x + hv.y * wv.y + hv.z * wv.z + hv.w * wv.w;
      }
      out[((size_t)t * BATCH + b0 + r) * ACT + a] = accp;
    }
    // no barrier needed: next phase writes only xbuf; h2f rewritten after 4 barriers
  }
}

extern "C" void kernel_launch(void* const* d_in, const int* in_sizes, int n_in,
                              void* d_out, int out_size, void* d_ws, size_t ws_size,
                              hipStream_t stream) {
  const float* x     = (const float*)d_in[0];
  const float* W_ih1 = (const float*)d_in[1];
  const float* W_hh1 = (const float*)d_in[2];
  const float* b_ih1 = (const float*)d_in[3];
  const float* b_hh1 = (const float*)d_in[4];
  const float* W_ih2 = (const float*)d_in[5];
  const float* W_hh2 = (const float*)d_in[6];
  const float* b_ih2 = (const float*)d_in[7];
  const float* b_hh2 = (const float*)d_in[8];
  const float* W_out = (const float*)d_in[9];
  const float* b_out = (const float*)d_in[10];
  float* out = (float*)d_out;

  // bf16 weight images in workspace (re-generated every call: deterministic)
  __bf16* wih1 = (__bf16*)d_ws;                 // 512*64
  __bf16* whh1 = wih1 + NGATE * S_IN;           // 512*128
  __bf16* wih2 = whh1 + NGATE * HID;            // 512*128
  __bf16* whh2 = wih2 + NGATE * HID;            // 512*128

  cvt_f32_bf16<<<(NGATE * S_IN + CVT_THREADS - 1) / CVT_THREADS, CVT_THREADS, 0,
                 stream>>>(W_ih1, wih1, NGATE * S_IN);
  cvt_f32_bf16<<<(NGATE * HID + CVT_THREADS - 1) / CVT_THREADS, CVT_THREADS, 0,
                 stream>>>(W_hh1, whh1, NGATE * HID);
  cvt_f32_bf16<<<(NGATE * HID + CVT_THREADS - 1) / CVT_THREADS, CVT_THREADS, 0,
                 stream>>>(W_ih2, wih2, NGATE * HID);
  cvt_f32_bf16<<<(NGATE * HID + CVT_THREADS - 1) / CVT_THREADS, CVT_THREADS, 0,
                 stream>>>(W_hh2, whh2, NGATE * HID);

  lstm2_wmma<<<BATCH / MTILE, NTHREADS, 0, stream>>>(
      x, wih1, whh1, wih2, whh2,
      b_ih1, b_hh1, b_ih2, b_hh2, W_out, b_out, out);
}
